// DetrCriterion_21423296872598
// MI455X (gfx1250) — matrix operand order repaired and checked by reading, not compiled, not run on hardware
//
#include <hip/hip_runtime.h>
#include <hip/hip_bf16.h>
#include <math.h>

// Problem constants (match reference)
#define Qn 900
#define Bn 256
#define C1n 92
#define Tn 64
#define NUM_CLASSES 91
#define INFV 3.0e38f

typedef __attribute__((ext_vector_type(16))) _Float16 v16h;
typedef __attribute__((ext_vector_type(8)))  float    v8f;

// ---------------------------------------------------------------------------
// Kernel 1: per-(q,b) log-sum-exp over 92 classes. One wave32 per row.
// lse[q*B + b] = max + log(sum exp(logit - max))
// ---------------------------------------------------------------------------
__global__ void lse_kernel(const float* __restrict__ logits,
                           float* __restrict__ lse) {
  const int gtid = blockIdx.x * blockDim.x + threadIdx.x;
  const int row  = gtid >> 5;              // (q*B + b)
  const int lane = threadIdx.x & 31;
  if (row >= Qn * Bn) return;
  const float* p = logits + (size_t)row * C1n;

  const float NEG = -3.0e38f;
  float v0 = p[lane];
  float v1 = p[lane + 32];
  float v2 = (lane + 64 < C1n) ? p[lane + 64] : NEG;
  float m = fmaxf(v0, fmaxf(v1, v2));
  #pragma unroll
  for (int off = 16; off >= 1; off >>= 1)
    m = fmaxf(m, __shfl_xor(m, off, 32));

  float e = __expf(v0 - m) + __expf(v1 - m) +
            ((lane + 64 < C1n) ? __expf(v2 - m) : 0.0f);
  #pragma unroll
  for (int off = 16; off >= 1; off >>= 1)
    e += __shfl_xor(e, off, 32);

  if (lane == 0) lse[row] = m + __logf(e);
}

// ---------------------------------------------------------------------------
// Kernel 2: cost matrix C[b][q][t] via WMMA for the class-gather term,
// fused with L1 + GIoU in the epilogue. One wave per 16-query tile.
//   cost_cls = probs(16x96,f16) x onehot(96x64)  (3 K-steps x 4 N-tiles)
// ---------------------------------------------------------------------------
__global__ void cost_wmma_kernel(const float* __restrict__ logits,
                                 const float* __restrict__ boxes,
                                 const int*   __restrict__ labels,
                                 const float* __restrict__ tboxes,
                                 const float* __restrict__ lse,
                                 float* __restrict__ Cmat) {
  __shared__ int   sLab[Tn];
  __shared__ float sTB[Tn * 4];

  const int b   = blockIdx.y;
  const int tid = threadIdx.x;
  if (tid < Tn)     sLab[tid] = labels[b * Tn + tid];
  if (tid < Tn * 4) sTB[tid]  = tboxes[(size_t)b * Tn * 4 + tid];
  __syncthreads();

  const int wave = tid >> 5;
  const int lane = tid & 31;
  const int tile = blockIdx.x * 8 + wave;          // 16-query tile index
  const int NTILES = (Qn + 15) / 16;               // 57
  if (tile >= NTILES) return;

  const int qbase = tile * 16;
  const int qa    = qbase + (lane & 15);           // A-frag row (M = lane&15)
  const int hiKA  = (lane >= 16) ? 16 : 0;         // A K-offset for hi lanes
  const int nB    = lane & 15;                     // B-frag column
  const int kB16  = (lane >= 16) ? 16 : 0;         // B K-offset for hi lanes

  // Hoist per-lane row constants for the A fragment.
  const bool   qok  = (qa < Qn);
  const size_t rb   = (size_t)(qok ? qa : 0) * Bn + b;
  const float  lrow = qok ? lse[rb] : 0.0f;
  const float* lptr = logits + rb * C1n;

  v8f acc[4] = {};

  #pragma unroll
  for (int kb = 0; kb < 3; ++kb) {                 // K = 96 in steps of 32
    const int kbase = kb * 32;
    // A fragment: probs (f16), contiguous k order (k = 16*hi + h), same as B.
    v16h a;
    #pragma unroll
    for (int h = 0; h < 16; ++h) {
      const int K = h + hiKA;                      // contiguous: k = 16*hi + h
      const int c = kbase + K;
      float v = 0.0f;
      if (qok && c < C1n) v = __expf(lptr[c] - lrow);
      a[h] = (_Float16)v;
    }
    // B fragments: one-hot(label). At most one non-zero half per lane, so
    // build through a u32 view: 8 compares + 8 dword selects per fragment.
    #pragma unroll
    for (int nt = 0; nt < 4; ++nt) {
      const int t   = nt * 16 + nB;
      const int p   = sLab[t] - kbase - kB16;      // position in this frag
      const unsigned hv = 0x3C00u << ((p & 1) * 16);
      union { v16h v; unsigned u[8]; } bu;
      #pragma unroll
      for (int r = 0; r < 8; ++r)
        bu.u[r] = ((p >> 1) == r) ? hv : 0u;       // p<0 or p>=16 -> no match
      acc[nt] = __builtin_amdgcn_wmma_f32_16x16x32_f16(
          false, a, false, bu.v, (short)0, acc[nt], false, false);
    }
  }

  // Epilogue: fuse L1 + GIoU.  D layout: VGPR r -> M = r + 8*(lane>=16),
  // lane -> N = lane&15.
  const int rq0 = qbase + ((lane >> 4) << 3);
  #pragma unroll
  for (int nt = 0; nt < 4; ++nt) {
    const int t = nt * 16 + (lane & 15);
    const float tcx = sTB[t * 4 + 0], tcy = sTB[t * 4 + 1];
    const float tw  = sTB[t * 4 + 2], tth = sTB[t * 4 + 3];
    const float tx0 = tcx - 0.5f * tw, ty0 = tcy - 0.5f * tth;
    const float tx1 = tcx + 0.5f * tw, ty1 = tcy + 0.5f * tth;
    const float ta  = tw * tth;
    #pragma unroll
    for (int r = 0; r < 8; ++r) {
      const int q = rq0 + r;
      if (q >= Qn) continue;
      const float* pb = boxes + ((size_t)q * Bn + b) * 4;
      const float cx = pb[0], cy = pb[1], w = pb[2], hh = pb[3];
      const float l1 = fabsf(cx - tcx) + fabsf(cy - tcy) +
                       fabsf(w - tw)   + fabsf(hh - tth);
      const float px0 = cx - 0.5f * w,  py0 = cy - 0.5f * hh;
      const float px1 = cx + 0.5f * w,  py1 = cy + 0.5f * hh;
      const float pa  = w * hh;
      const float iw = fmaxf(fminf(px1, tx1) - fmaxf(px0, tx0), 0.0f);
      const float ih = fmaxf(fminf(py1, ty1) - fmaxf(py0, ty0), 0.0f);
      const float inter = iw * ih;
      const float uni   = pa + ta - inter;
      const float iou   = inter / uni;
      const float ew = fmaxf(px1, tx1) - fminf(px0, tx0);
      const float eh = fmaxf(py1, ty1) - fminf(py0, ty0);
      const float enc = ew * eh;
      const float g   = iou - (enc - uni) / enc;
      const float cost = -acc[nt][r] + 5.0f * l1 - 2.0f * g;
      Cmat[((size_t)b * Qn + q) * Tn + t] = cost;
    }
  }
}

// ---------------------------------------------------------------------------
// Kernel 3: greedy assignment with incremental per-row minima.
// One block per batch item. rmv/rmc cache each row's (minval, mincol); a
// step only rescans rows whose cached min column was just masked.
// Tie-breaks replicate jnp.argmin's flat row-major first occurrence:
// strict-< ascending scans (in-row: lower t; global: lower q).
// ---------------------------------------------------------------------------
__global__ void greedy_kernel(const float* __restrict__ Cmat,
                              const int*   __restrict__ labels,
                              int* __restrict__ tcls,
                              int* __restrict__ ii,
                              int* __restrict__ jj) {
  __shared__ float rmv[Qn];       // per-row min value (INFV if row used)
  __shared__ int   rmc[Qn];       // per-row min column
  __shared__ int   colUsed[Tn];
  __shared__ int   sLab[Tn];
  __shared__ float rv[256];
  __shared__ int   ri[256];
  __shared__ int   sTJ;

  const int b   = blockIdx.x;
  const int tid = threadIdx.x;
  const float* Cb = Cmat + (size_t)b * Qn * Tn;

  // Init: per-row minima (first-occurrence on ties) + tgt_class defaults.
  for (int q = tid; q < Qn; q += 256) {
    const float* row = Cb + (size_t)q * Tn;
    float nv = INFV; int nc = 0;
    for (int t = 0; t < Tn; ++t) {
      __builtin_prefetch(&row[(t + 16 < Tn) ? t + 16 : t], 0, 1);
      const float v = row[t];
      if (v < nv) { nv = v; nc = t; }
    }
    rmv[q] = nv; rmc[q] = nc;
    tcls[b * Qn + q] = NUM_CLASSES;
  }
  if (tid < Tn) { colUsed[tid] = 0; sLab[tid] = labels[b * Tn + tid]; }
  __syncthreads();

  for (int s = 0; s < Tn; ++s) {
    // Global argmin over row minima (tie -> lower q).
    float bv = INFV; int bq = 0x7fffffff;
    for (int q = tid; q < Qn; q += 256) {
      const float v = rmv[q];
      if (v < bv) { bv = v; bq = q; }   // strided ascending: first min kept
    }
    rv[tid] = bv; ri[tid] = bq;
    __syncthreads();
    #pragma unroll
    for (int off = 128; off >= 1; off >>= 1) {
      if (tid < off) {
        const float ov = rv[tid + off];
        const int   oq = ri[tid + off];
        if (ov < rv[tid] || (ov == rv[tid] && oq < ri[tid])) {
          rv[tid] = ov; ri[tid] = oq;
        }
      }
      __syncthreads();
    }
    if (tid == 0) {
      const int pi = ri[0];
      const int tj = rmc[pi];
      sTJ = tj;
      rmv[pi] = INFV;                 // row consumed
      colUsed[tj] = 1;
      ii[b * Tn + s] = pi;
      jj[b * Tn + s] = tj;
      tcls[b * Qn + pi] = sLab[tj];
    }
    __syncthreads();

    // Rescan only rows whose cached min column was just masked.
    const int tj = sTJ;
    for (int q = tid; q < Qn; q += 256) {
      if (rmv[q] < INFV && rmc[q] == tj) {
        const float* row = Cb + (size_t)q * Tn;
        float nv = INFV; int nc = 0;
        for (int t = 0; t < Tn; ++t) {
          if (colUsed[t]) continue;
          const float v = row[t];
          if (v < nv) { nv = v; nc = t; }
        }
        rmv[q] = nv; rmc[q] = nc;
      }
    }
    __syncthreads();
  }
}

// ---------------------------------------------------------------------------
// Kernel 4: CE partial sums.  -logp[b,q,c] = lse[q,b] - logit[q,b,c]
// ---------------------------------------------------------------------------
__global__ void ce_kernel(const float* __restrict__ logits,
                          const float* __restrict__ lse,
                          const int*   __restrict__ tcls,
                          float* __restrict__ part) {
  __shared__ float red[256];
  const int tid = threadIdx.x;
  float s = 0.0f;
  for (int i = blockIdx.x * 256 + tid; i < Bn * Qn; i += gridDim.x * 256) {
    const int b = i / Qn, q = i - b * Qn;
    const int c = tcls[i];
    const size_t rb = (size_t)q * Bn + b;
    s += lse[rb] - logits[rb * C1n + c];
  }
  red[tid] = s;
  __syncthreads();
  #pragma unroll
  for (int off = 128; off >= 1; off >>= 1) {
    if (tid < off) red[tid] += red[tid + off];
    __syncthreads();
  }
  if (tid == 0) part[blockIdx.x] = red[0];
}

// ---------------------------------------------------------------------------
// Kernel 5: matched-box L1 and GIoU partial sums over B*T pairs.
// ---------------------------------------------------------------------------
__global__ void box_kernel(const float* __restrict__ boxes,
                           const float* __restrict__ tboxes,
                           const int* __restrict__ ii,
                           const int* __restrict__ jj,
                           float* __restrict__ partL1,
                           float* __restrict__ partG) {
  __shared__ float r1[256];
  __shared__ float r2[256];
  const int tid = threadIdx.x;
  float sl = 0.0f, sg = 0.0f;
  for (int i = blockIdx.x * 256 + tid; i < Bn * Tn; i += gridDim.x * 256) {
    const int b = i / Tn;
    const int q = ii[i];
    const int t = jj[i];
    const float* pb = boxes  + ((size_t)q * Bn + b) * 4;
    const float* tb = tboxes + ((size_t)b * Tn + t) * 4;
    const float cx = pb[0], cy = pb[1], w = pb[2], hh = pb[3];
    const float tcx = tb[0], tcy = tb[1], tw = tb[2], tth = tb[3];
    sl += fabsf(cx - tcx) + fabsf(cy - tcy) + fabsf(w - tw) + fabsf(hh - tth);
    const float px0 = cx - 0.5f * w,  py0 = cy - 0.5f * hh;
    const float px1 = cx + 0.5f * w,  py1 = cy + 0.5f * hh;
    const float tx0 = tcx - 0.5f * tw, ty0 = tcy - 0.5f * tth;
    const float tx1 = tcx + 0.5f * tw, ty1 = tcy + 0.5f * tth;
    const float pa = w * hh, ta = tw * tth;
    const float iw = fmaxf(fminf(px1, tx1) - fmaxf(px0, tx0), 0.0f);
    const float ih = fmaxf(fminf(py1, ty1) - fmaxf(py0, ty0), 0.0f);
    const float inter = iw * ih;
    const float uni = pa + ta - inter;
    const float iou = inter / uni;
    const float ew = fmaxf(px1, tx1) - fminf(px0, tx0);
    const float eh = fmaxf(py1, ty1) - fminf(py0, ty0);
    const float enc = ew * eh;
    sg += iou - (enc - uni) / enc;
  }
  r1[tid] = sl; r2[tid] = sg;
  __syncthreads();
  #pragma unroll
  for (int off = 128; off >= 1; off >>= 1) {
    if (tid < off) { r1[tid] += r1[tid + off]; r2[tid] += r2[tid + off]; }
    __syncthreads();
  }
  if (tid == 0) { partL1[blockIdx.x] = r1[0]; partG[blockIdx.x] = r2[0]; }
}

// ---------------------------------------------------------------------------
// Kernel 6: final combine -> out = [loss, ce, l1, giou_loss]
// ---------------------------------------------------------------------------
__global__ void final_kernel(const float* __restrict__ pce, int nce,
                             const float* __restrict__ pl1,
                             const float* __restrict__ pg, int nb,
                             float* __restrict__ out) {
  __shared__ float ra[256];
  __shared__ float rb_[256];
  __shared__ float rc[256];
  const int tid = threadIdx.x;
  float a = 0.0f, bsum = 0.0f, c = 0.0f;
  for (int i = tid; i < nce; i += 256) a += pce[i];
  for (int i = tid; i < nb;  i += 256) { bsum += pl1[i]; c += pg[i]; }
  ra[tid] = a; rb_[tid] = bsum; rc[tid] = c;
  __syncthreads();
  #pragma unroll
  for (int off = 128; off >= 1; off >>= 1) {
    if (tid < off) {
      ra[tid] += ra[tid + off];
      rb_[tid] += rb_[tid + off];
      rc[tid] += rc[tid + off];
    }
    __syncthreads();
  }
  if (tid == 0) {
    const float ce = ra[0] / (float)(Bn * Qn);
    const float l1 = rb_[0] / (float)(Bn * Tn * 4);
    const float gl = 1.0f - rc[0] / (float)(Bn * Tn);
    out[0] = 1.0f * ce + 5.0f * l1 + 2.0f * gl;
    out[1] = ce;
    out[2] = l1;
    out[3] = gl;
  }
}

// ---------------------------------------------------------------------------
// Host launcher
// ---------------------------------------------------------------------------
extern "C" void kernel_launch(void* const* d_in, const int* in_sizes, int n_in,
                              void* d_out, int out_size, void* d_ws, size_t ws_size,
                              hipStream_t stream) {
  const float* d_logits = (const float*)d_in[0];   // (Q,B,C1)
  const float* d_boxes  = (const float*)d_in[1];   // (Q,B,4)
  const int*   d_labels = (const int*)  d_in[2];   // (B,T)
  const float* d_tboxes = (const float*)d_in[3];   // (B,T,4)
  float* out = (float*)d_out;

  const int NCE = 512;   // CE partial blocks
  const int NBX = 64;    // box partial blocks

  // Workspace layout
  char* p = (char*)d_ws;
  size_t o = 0;
  float* d_lse  = (float*)(p + o); o += (size_t)Qn * Bn * sizeof(float);
  float* d_C    = (float*)(p + o); o += (size_t)Bn * Qn * Tn * sizeof(float);
  int*   d_tcls = (int*)  (p + o); o += (size_t)Bn * Qn * sizeof(int);
  int*   d_ii   = (int*)  (p + o); o += (size_t)Bn * Tn * sizeof(int);
  int*   d_jj   = (int*)  (p + o); o += (size_t)Bn * Tn * sizeof(int);
  float* d_pce  = (float*)(p + o); o += (size_t)NCE * sizeof(float);
  float* d_pl1  = (float*)(p + o); o += (size_t)NBX * sizeof(float);
  float* d_pg   = (float*)(p + o); o += (size_t)NBX * sizeof(float);
  (void)ws_size; (void)n_in; (void)in_sizes; (void)out_size;

  // 1) log-sum-exp per (q,b): one wave per row
  {
    const int rows = Qn * Bn;
    const int blocks = (rows * 32 + 255) / 256;
    lse_kernel<<<blocks, 256, 0, stream>>>(d_logits, d_lse);
  }
  // 2) cost matrix (WMMA class gather + fused L1/GIoU)
  {
    dim3 grid(8, Bn);   // 8 blocks x 8 waves = 64 wave-tiles >= 57 needed
    cost_wmma_kernel<<<grid, 256, 0, stream>>>(d_logits, d_boxes, d_labels,
                                               d_tboxes, d_lse, d_C);
  }
  // 3) greedy assignment (one block per batch item, incremental row minima)
  greedy_kernel<<<Bn, 256, 0, stream>>>(d_C, d_labels, d_tcls, d_ii, d_jj);
  // 4) CE partials
  ce_kernel<<<NCE, 256, 0, stream>>>(d_logits, d_lse, d_tcls, d_pce);
  // 5) matched-box L1/GIoU partials
  box_kernel<<<NBX, 256, 0, stream>>>(d_boxes, d_tboxes, d_ii, d_jj, d_pl1, d_pg);
  // 6) combine
  final_kernel<<<1, 256, 0, stream>>>(d_pce, NCE, d_pl1, d_pg, NBX, out);
}
